// EGNNLSPE_88167088653030
// MI455X (gfx1250) — matrix operand order repaired and checked by reading, not compiled, not run on hardware
//
#include <hip/hip_runtime.h>
#include <math.h>

typedef __attribute__((ext_vector_type(16))) _Float16 v16h;
typedef __attribute__((ext_vector_type(8)))  float    v8f;

#define HDIM 96
#define NTL  6   // HDIM / 16

__device__ __forceinline__ float fatomic_add(float* p, float v) {
  return __hip_atomic_fetch_add(p, v, __ATOMIC_RELAXED, __HIP_MEMORY_SCOPE_AGENT);
}

// ---------------------------------------------------------------------------
// Fragment-native storage (ISA 7.12.2, wave32):
//  A 16x32 f16 : lane L -> M=L%16 ; elem j -> K = (j%8) + (L/16)*8 + (j/8)*16
//  B 32x16 f16 : lane L -> N=n0+L%16 ; elem j -> K = k0 + (j%8) + (L/16)*8 + (j/8)*16
// We pre-pack both A (in LDS) and B (in global ws) so a fragment is one
// contiguous 32-byte per-lane vector -> b128 pair loads, no guards/cvt.
// Packed layout: frag_base = tile_index*32*16 ; lane slice = lane*16 f16.
// ---------------------------------------------------------------------------

// scatter one A element (row r, column c of the logical A tile) into packed LDS
__device__ __forceinline__ void stash_a(_Float16* Asw, int r, int c, float v) {
  const int kt   = c >> 5;
  const int kc   = c & 31;
  const int g    = kc >> 3;                 // quarter of the k-tile
  const int lane = r + ((g & 1) << 4);
  const int j    = (kc & 7) + ((g >> 1) << 3);
  Asw[(kt << 9) + (lane << 4) + j] = (_Float16)v;
}

__device__ __forceinline__ v16h load_a_sw(const _Float16* Asw, int kt, int lane) {
  return *(const v16h*)(Asw + (kt << 9) + (lane << 4));
}

__device__ __forceinline__ v16h load_b_sw(const _Float16* Wsw, int kt, int nt, int lane) {
  return *(const v16h*)(Wsw + ((((size_t)(kt * 6 + nt)) << 5) + lane) * 16);
}

// one-time weight pre-pack: grid = nkt*6 blocks of 32 lanes
__global__ __launch_bounds__(32)
void swizzle_w_kernel(const float* __restrict__ W, int Ktot, _Float16* __restrict__ out) {
  const int lane = threadIdx.x;
  const int kt = blockIdx.x / 6, nt = blockIdx.x % 6;
  const int n  = nt * 16 + (lane & 15);
  const int kb = kt * 32 + ((lane >> 4) << 3);
  _Float16* p = out + ((((size_t)blockIdx.x) << 5) + lane) * 16;
#pragma unroll
  for (int j = 0; j < 16; ++j) {
    int k = kb + (j & 7) + ((j >> 3) << 4);
    float v = (k < Ktot) ? W[k * HDIM + n] : 0.0f;
    p[j] = (_Float16)v;
  }
}

// ---------------------------------------------------------------------------
// Generic two-layer MLP: out = relu(A @ W1 + B1) @ W2 + B2
// A rows assembled from up to 3 row-major segments (s0|s1|s2).
// One wave handles a 16-row tile; K <= 288 (9 k-tiles).
// ---------------------------------------------------------------------------
__global__ __launch_bounds__(32)
void mlp2_wmma_kernel(const float* __restrict__ s0, int w0, int ld0,
                      const float* __restrict__ s1, int w1, int ld1,
                      const float* __restrict__ s2, int w2, int ld2,
                      int rows, int K,
                      const _Float16* __restrict__ W1sw, const float* __restrict__ B1,
                      const _Float16* __restrict__ W2sw, const float* __restrict__ B2,
                      float* __restrict__ out)
{
  __shared__ __align__(32) _Float16 Asw[9 * 512];
  __shared__ __align__(32) _Float16 Msw[3 * 512];
  const int lane = threadIdx.x;
  const int r0   = blockIdx.x * 16;
  const int Kpad = (K + 31) & ~31;

  for (int r = 0; r < 16; ++r) {
    int  row = r0 + r;
    bool ok  = row < rows;
    for (int c = lane; c < Kpad; c += 32) {
      float v = 0.0f;
      if (ok) {
        if (c < w0)           v = s0[row * ld0 + c];
        else if (c < w0 + w1) v = s1[row * ld1 + (c - w0)];
        else if (c < K)       v = s2[row * ld2 + (c - w0 - w1)];
      }
      stash_a(Asw, r, c, v);
    }
  }
  __syncthreads();

  v8f acc[NTL];
#pragma unroll
  for (int t = 0; t < NTL; ++t)
#pragma unroll
    for (int r = 0; r < 8; ++r) acc[t][r] = 0.0f;

  const int nkt = Kpad >> 5;
  for (int kt = 0; kt < nkt; ++kt) {
    v16h a = load_a_sw(Asw, kt, lane);
#pragma unroll
    for (int t = 0; t < NTL; ++t) {
      v16h b = load_b_sw(W1sw, kt, t, lane);
      acc[t] = __builtin_amdgcn_wmma_f32_16x16x32_f16(false, a, false, b,
                                                      (short)0, acc[t], false, false);
    }
  }

  // bias + relu -> packed Msw
  {
    const int n_lo = lane & 15;
    const int mh   = (lane >> 4) << 3;
#pragma unroll
    for (int t = 0; t < NTL; ++t) {
      int   n   = t * 16 + n_lo;
      float bia = B1[n];
#pragma unroll
      for (int r = 0; r < 8; ++r) {
        float v = acc[t][r] + bia;
        stash_a(Msw, mh + r, n, v > 0.0f ? v : 0.0f);
      }
    }
  }
  __syncthreads();

#pragma unroll
  for (int t = 0; t < NTL; ++t)
#pragma unroll
    for (int r = 0; r < 8; ++r) acc[t][r] = 0.0f;

#pragma unroll
  for (int kt = 0; kt < 3; ++kt) {
    v16h a = load_a_sw(Msw, kt, lane);
#pragma unroll
    for (int t = 0; t < NTL; ++t) {
      v16h b = load_b_sw(W2sw, kt, t, lane);
      acc[t] = __builtin_amdgcn_wmma_f32_16x16x32_f16(false, a, false, b,
                                                      (short)0, acc[t], false, false);
    }
  }

  {
    const int n_lo = lane & 15;
    const int mh   = (lane >> 4) << 3;
#pragma unroll
    for (int t = 0; t < NTL; ++t) {
      int   n   = t * 16 + n_lo;
      float bia = B2[n];
#pragma unroll
      for (int r = 0; r < 8; ++r) {
        int row = r0 + mh + r;
        if (row < rows) out[row * HDIM + n] = acc[t][r] + bia;
      }
    }
  }
}

// ---------------------------------------------------------------------------
// Edge message kernel: 16 edges per wave.
// m_in = [h[dst], h[src], pe[dst], pe[src], dist]  (K=385, padded to 416)
// m = relu(m_in@W1+B1); m = relu(m@W2+B2); m *= sigmoid(m@GW+GB)
// scatter-add into agg[dst] (mean division happens afterwards).
// ---------------------------------------------------------------------------
__global__ __launch_bounds__(32)
void edge_msg_kernel(const float* __restrict__ h, const float* __restrict__ pe,
                     const float* __restrict__ dist,
                     const int* __restrict__ srci, const int* __restrict__ dsti,
                     const _Float16* __restrict__ W1sw, const float* __restrict__ B1,
                     const _Float16* __restrict__ W2sw, const float* __restrict__ B2,
                     const float* __restrict__ GW, const float* __restrict__ GB,
                     float* __restrict__ agg, int E)
{
  __shared__ __align__(32) _Float16 Asw[13 * 512];
  __shared__ float M2[16][HDIM];
  __shared__ float gs[16];
  __shared__ int   didx[16];
  const int lane = threadIdx.x;
  const int e0   = blockIdx.x * 16;

  if (lane < 16) {
    int e = e0 + lane;
    didx[lane] = (e < E) ? dsti[e] : -1;
  }

  for (int r = 0; r < 16; ++r) {
    int e = e0 + r;
    if (e < E) {
      int s = srci[e], d = dsti[e];
      const float* hd = h  + d * HDIM;
      const float* hs = h  + s * HDIM;
      const float* pd = pe + d * HDIM;
      const float* ps = pe + s * HDIM;
#pragma unroll
      for (int it = 0; it < 3; ++it) {
        int c = lane + it * 32;
        stash_a(Asw, r, c,       hd[c]);
        stash_a(Asw, r, 96 + c,  hs[c]);
        stash_a(Asw, r, 192 + c, pd[c]);
        stash_a(Asw, r, 288 + c, ps[c]);
      }
      int c = 384 + lane;                          // lanes cover 384..415
      stash_a(Asw, r, c, (c == 384) ? dist[e] : 0.0f);
    } else {
      for (int c = lane; c < 416; c += 32) stash_a(Asw, r, c, 0.0f);
    }
  }
  __syncthreads();

  v8f acc[NTL];
#pragma unroll
  for (int t = 0; t < NTL; ++t)
#pragma unroll
    for (int r = 0; r < 8; ++r) acc[t][r] = 0.0f;

#pragma unroll
  for (int kt = 0; kt < 13; ++kt) {                // K=385 -> 13 k-tiles
    v16h a = load_a_sw(Asw, kt, lane);
#pragma unroll
    for (int t = 0; t < NTL; ++t) {
      v16h b = load_b_sw(W1sw, kt, t, lane);
      acc[t] = __builtin_amdgcn_wmma_f32_16x16x32_f16(false, a, false, b,
                                                      (short)0, acc[t], false, false);
    }
  }
  __syncthreads();                                 // done reading Asw

  // bias + relu -> reuse first 3 k-tiles of Asw as packed m1
  {
    const int n_lo = lane & 15;
    const int mh   = (lane >> 4) << 3;
#pragma unroll
    for (int t = 0; t < NTL; ++t) {
      int   n   = t * 16 + n_lo;
      float bia = B1[n];
#pragma unroll
      for (int r = 0; r < 8; ++r) {
        float v = acc[t][r] + bia;
        stash_a(Asw, mh + r, n, v > 0.0f ? v : 0.0f);
      }
    }
  }
  __syncthreads();

#pragma unroll
  for (int t = 0; t < NTL; ++t)
#pragma unroll
    for (int r = 0; r < 8; ++r) acc[t][r] = 0.0f;

#pragma unroll
  for (int kt = 0; kt < 3; ++kt) {
    v16h a = load_a_sw(Asw, kt, lane);
#pragma unroll
    for (int t = 0; t < NTL; ++t) {
      v16h b = load_b_sw(W2sw, kt, t, lane);
      acc[t] = __builtin_amdgcn_wmma_f32_16x16x32_f16(false, a, false, b,
                                                      (short)0, acc[t], false, false);
    }
  }

  {
    const int n_lo = lane & 15;
    const int mh   = (lane >> 4) << 3;
#pragma unroll
    for (int t = 0; t < NTL; ++t) {
      int   n   = t * 16 + n_lo;
      float bia = B2[n];
#pragma unroll
      for (int r = 0; r < 8; ++r) {
        float v = acc[t][r] + bia;
        M2[mh + r][n] = v > 0.0f ? v : 0.0f;
      }
    }
  }
  __syncthreads();

  if (lane < 16) {
    float g = GB[0];
    for (int n = 0; n < HDIM; ++n) g += M2[lane][n] * GW[n];
    gs[lane] = 1.0f / (1.0f + expf(-g));
  }
  __syncthreads();

  for (int r = 0; r < 16; ++r) {
    int d = didx[r];
    if (d >= 0) {
      float sc = gs[r];
#pragma unroll
      for (int it = 0; it < 3; ++it) {
        int c = lane + it * 32;
        fatomic_add(&agg[d * HDIM + c], M2[r][c] * sc);
      }
    }
  }
}

// ---- small helper kernels --------------------------------------------------
__global__ void zero_kernel(float* p, int n) {
  int i = blockIdx.x * blockDim.x + threadIdx.x;
  if (i < n) p[i] = 0.0f;
}

__global__ void edge_pre_kernel(const float* __restrict__ pos,
                                const int* __restrict__ srci, const int* __restrict__ dsti,
                                float* __restrict__ dist, float* __restrict__ deg, int E) {
  int e = blockIdx.x * blockDim.x + threadIdx.x;
  if (e >= E) return;
  int s = srci[e], d = dsti[e];
  float dx = pos[d * 3 + 0] - pos[s * 3 + 0];
  float dy = pos[d * 3 + 1] - pos[s * 3 + 1];
  float dz = pos[d * 3 + 2] - pos[s * 3 + 2];
  dist[e] = sqrtf(dx * dx + dy * dy + dz * dz);
  fatomic_add(&deg[d], 1.0f);
}

__global__ void invdeg_kernel(float* deg, int N) {
  int i = blockIdx.x * blockDim.x + threadIdx.x;
  if (i < N) deg[i] = 1.0f / fmaxf(deg[i], 1.0f);
}

// agg[n][:] *= invdeg[n]  (turns the scatter-sum into a mean)
__global__ void scale_rows_kernel(float* __restrict__ a, const float* __restrict__ s, int total) {
  int i = blockIdx.x * blockDim.x + threadIdx.x;
  if (i < total) a[i] *= s[i / HDIM];
}

__global__ void pe_embed_kernel(const float* __restrict__ pe_init,
                                const float* __restrict__ W, const float* __restrict__ B,
                                float* __restrict__ pe, int N) {
  int i = blockIdx.x * blockDim.x + threadIdx.x;
  if (i >= N * HDIM) return;
  int n = i / HDIM, c = i % HDIM;
  float v = B[c];
#pragma unroll
  for (int k = 0; k < 8; ++k) v += pe_init[n * 8 + k] * W[k * HDIM + c];
  pe[i] = v;
}

__global__ void pool_kernel(const float* __restrict__ hfin, const int* __restrict__ batch,
                            float* __restrict__ pooled, int N) {
  int i = blockIdx.x * blockDim.x + threadIdx.x;
  if (i >= N * HDIM) return;
  int n = i / HDIM, c = i % HDIM;
  fatomic_add(&pooled[batch[n] * HDIM + c], hfin[i]);
}

// ---------------------------------------------------------------------------
extern "C" void kernel_launch(void* const* d_in, const int* in_sizes, int n_in,
                              void* d_out, int out_size, void* d_ws, size_t ws_size,
                              hipStream_t stream) {
  const float* x       = (const float*)d_in[0];
  const float* pos     = (const float*)d_in[1];
  const float* pe_init = (const float*)d_in[2];
  const int*   eidx    = (const int*)d_in[3];
  const int*   batch   = (const int*)d_in[4];
  const float* ex_w1 = (const float*)d_in[5],  *ex_b1 = (const float*)d_in[6];
  const float* ex_w2 = (const float*)d_in[7],  *ex_b2 = (const float*)d_in[8];
  const float* ep_w  = (const float*)d_in[9],  *ep_b  = (const float*)d_in[10];
  const float* msg_w1 = (const float*)d_in[11], *msg_b1 = (const float*)d_in[12];
  const float* msg_w2 = (const float*)d_in[13], *msg_b2 = (const float*)d_in[14];
  const float* gate_w = (const float*)d_in[15], *gate_b = (const float*)d_in[16];
  const float* ux_w1 = (const float*)d_in[17], *ux_b1 = (const float*)d_in[18];
  const float* ux_w2 = (const float*)d_in[19], *ux_b2 = (const float*)d_in[20];
  const float* up_w1 = (const float*)d_in[21], *up_b1 = (const float*)d_in[22];
  const float* up_w2 = (const float*)d_in[23], *up_b2 = (const float*)d_in[24];
  const float* nd_w1 = (const float*)d_in[25], *nd_b1 = (const float*)d_in[26];
  const float* nd_w2 = (const float*)d_in[27], *nd_b2 = (const float*)d_in[28];
  const float* hd_w1 = (const float*)d_in[29], *hd_b1 = (const float*)d_in[30];
  const float* hd_w2 = (const float*)d_in[31], *hd_b2 = (const float*)d_in[32];

  const int N = in_sizes[0] / 11;
  const int E = in_sizes[3] / 2;
  const int G = out_size / HDIM;
  const int* srci = eidx;
  const int* dsti = eidx + E;

  float* ws = (float*)d_ws;
  float* h      = ws; ws += (size_t)N * HDIM;
  float* pe     = ws; ws += (size_t)N * HDIM;
  float* h2     = ws; ws += (size_t)N * HDIM;
  float* pe2    = ws; ws += (size_t)N * HDIM;
  float* agg    = ws; ws += (size_t)N * HDIM;   // reused as hfin at the end
  float* invdeg = ws; ws += (size_t)N;
  float* dist   = ws; ws += (size_t)E;
  float* pooled = ws; ws += (size_t)G * HDIM;

  // f16 swizzled-weight region (32B aligned)
  _Float16* wsw = (_Float16*)(((uintptr_t)ws + 31) & ~(uintptr_t)31);
  size_t off = 0;
  auto walloc = [&](int nkt) { size_t o = off; off += (size_t)nkt * 3072; return o; };
  size_t o_ex1 = walloc(1), o_ex2 = walloc(3);
  size_t o_nd1 = walloc(3), o_nd2 = walloc(3);
  size_t o_hd1 = walloc(3), o_hd2 = walloc(3);
  size_t o_m1[3], o_m2[3], o_x1[3], o_x2[3], o_p1[3], o_p2[3];
  for (int l = 0; l < 3; ++l) {
    o_m1[l] = walloc(13); o_m2[l] = walloc(3);
    o_x1[l] = walloc(9);  o_x2[l] = walloc(3);
    o_p1[l] = walloc(5);  o_p2[l] = walloc(3);
  }

  const int TB = 256;
  auto cdiv = [](int a, int b) { return (a + b - 1) / b; };

  // ---- pre-pack all weights to fragment-native f16 -------------------------
  swizzle_w_kernel<<<1 * 6, 32, 0, stream>>>(ex_w1, 11, wsw + o_ex1);
  swizzle_w_kernel<<<3 * 6, 32, 0, stream>>>(ex_w2, 96, wsw + o_ex2);
  swizzle_w_kernel<<<3 * 6, 32, 0, stream>>>(nd_w1, 96, wsw + o_nd1);
  swizzle_w_kernel<<<3 * 6, 32, 0, stream>>>(nd_w2, 96, wsw + o_nd2);
  swizzle_w_kernel<<<3 * 6, 32, 0, stream>>>(hd_w1, 96, wsw + o_hd1);
  swizzle_w_kernel<<<3 * 6, 32, 0, stream>>>(hd_w2, 96, wsw + o_hd2);
  for (int l = 0; l < 3; ++l) {
    swizzle_w_kernel<<<13 * 6, 32, 0, stream>>>(msg_w1 + (size_t)l * 385 * HDIM, 385, wsw + o_m1[l]);
    swizzle_w_kernel<<<3 * 6, 32, 0, stream>>>(msg_w2 + (size_t)l * HDIM * HDIM, 96, wsw + o_m2[l]);
    swizzle_w_kernel<<<9 * 6, 32, 0, stream>>>(ux_w1 + (size_t)l * 288 * HDIM, 288, wsw + o_x1[l]);
    swizzle_w_kernel<<<3 * 6, 32, 0, stream>>>(ux_w2 + (size_t)l * HDIM * HDIM, 96, wsw + o_x2[l]);
    swizzle_w_kernel<<<5 * 6, 32, 0, stream>>>(up_w1 + (size_t)l * 144 * HDIM, 144, wsw + o_p1[l]);
    swizzle_w_kernel<<<3 * 6, 32, 0, stream>>>(up_w2 + (size_t)l * HDIM * HDIM, 96, wsw + o_p2[l]);
  }

  // ---- embedders -----------------------------------------------------------
  mlp2_wmma_kernel<<<cdiv(N, 16), 32, 0, stream>>>(
      x, 11, 11, nullptr, 0, 0, nullptr, 0, 0,
      N, 11, wsw + o_ex1, ex_b1, wsw + o_ex2, ex_b2, h);
  pe_embed_kernel<<<cdiv(N * HDIM, TB), TB, 0, stream>>>(pe_init, ep_w, ep_b, pe, N);

  // ---- dist + in-degree -> invdeg ------------------------------------------
  zero_kernel<<<cdiv(N, TB), TB, 0, stream>>>(invdeg, N);
  edge_pre_kernel<<<cdiv(E, TB), TB, 0, stream>>>(pos, srci, dsti, dist, invdeg, E);
  invdeg_kernel<<<cdiv(N, TB), TB, 0, stream>>>(invdeg, N);

  // ---- message-passing layers ----------------------------------------------
  for (int l = 0; l < 3; ++l) {
    zero_kernel<<<cdiv(N * HDIM, TB), TB, 0, stream>>>(agg, N * HDIM);
    edge_msg_kernel<<<cdiv(E, 16), 32, 0, stream>>>(
        h, pe, dist, srci, dsti,
        wsw + o_m1[l], msg_b1 + (size_t)l * HDIM,
        wsw + o_m2[l], msg_b2 + (size_t)l * HDIM,
        gate_w + (size_t)l * HDIM, gate_b + l,
        agg, E);
    scale_rows_kernel<<<cdiv(N * HDIM, TB), TB, 0, stream>>>(agg, invdeg, N * HDIM);
    // h update: [h | pe | agg_mean]  (K=288)
    mlp2_wmma_kernel<<<cdiv(N, 16), 32, 0, stream>>>(
        h, HDIM, HDIM, pe, HDIM, HDIM, agg, HDIM, HDIM,
        N, 288,
        wsw + o_x1[l], ux_b1 + (size_t)l * HDIM,
        wsw + o_x2[l], ux_b2 + (size_t)l * HDIM, h2);
    // pe update: [pe | agg_mean[:, :48]]  (K=144)
    mlp2_wmma_kernel<<<cdiv(N, 16), 32, 0, stream>>>(
        pe, HDIM, HDIM, agg, 48, HDIM, nullptr, 0, 0,
        N, 144,
        wsw + o_p1[l], up_b1 + (size_t)l * HDIM,
        wsw + o_p2[l], up_b2 + (size_t)l * HDIM, pe2);
    float* t;
    t = h;  h  = h2;  h2  = t;
    t = pe; pe = pe2; pe2 = t;
  }

  // ---- node decoder -> hfin (reuse agg) ------------------------------------
  mlp2_wmma_kernel<<<cdiv(N, 16), 32, 0, stream>>>(
      h, HDIM, HDIM, nullptr, 0, 0, nullptr, 0, 0,
      N, HDIM, wsw + o_nd1, nd_b1, wsw + o_nd2, nd_b2, agg);

  // ---- global add pool + head ----------------------------------------------
  zero_kernel<<<cdiv(G * HDIM, TB), TB, 0, stream>>>(pooled, G * HDIM);
  pool_kernel<<<cdiv(N * HDIM, TB), TB, 0, stream>>>(agg, batch, pooled, N);
  mlp2_wmma_kernel<<<cdiv(G, 16), 32, 0, stream>>>(
      pooled, HDIM, HDIM, nullptr, 0, 0, nullptr, 0, 0,
      G, HDIM, wsw + o_hd1, hd_b1, wsw + o_hd2, hd_b2, (float*)d_out);
}
